// I2P_MAE_36515811950954
// MI455X (gfx1250) — compile-verified, matrix operand order checked by reading, not run on hardware
//
#include <hip/hip_runtime.h>
#include <hip/hip_bf16.h>
#include <math.h>

typedef __attribute__((ext_vector_type(16))) _Float16 v16h;
typedef __attribute__((ext_vector_type(8)))  _Float16 v8h;
typedef __attribute__((ext_vector_type(8)))  float    v8f;

#define GEPS 1e-5f
#define NHEADS 6

// ---------------------------------------------------------------------------
// Weight transpose + f32->f16 convert:  Wt[n*K + k] = (f16) W[k*N + n]
// Makes the B-fragment K-contiguous so the GEMM can use 128-bit loads.
// ---------------------------------------------------------------------------
__global__ void transpose_h_kernel(const float* __restrict__ W, _Float16* __restrict__ Wt,
                                   int K, int N)
{
    long t = (long)blockIdx.x * blockDim.x + threadIdx.x;
    if (t >= (long)K * N) return;
    int n = (int)(t % N); long k = t / N;
    Wt[(long)n * K + k] = (_Float16)W[t];
}

// ---------------------------------------------------------------------------
// WMMA GEMM:  C[M,N] = epilogue( A[M,K] @ W[K,N] ),  W given transposed f16.
// REQUIRES: M%16==0, N%16==0, and (K%32==0 or K<32); lda%4==0 when K>=32.
// EPI: 0=none 1=+bias 2=+bias,BN,ReLU 3=+bias,GELU(exact) 4=+bias,+residual
// One wave per 16x16 C tile, 8 waves/block, unguarded main loop (no exec
// divergence), branch-free clamped tail for K%32 remainder (only K=3 here).
// ---------------------------------------------------------------------------
template<int EPI>
__global__ __launch_bounds__(256)
void wmma_gemm_kernel(const float* __restrict__ A, int lda,
                      const _Float16* __restrict__ Wt,   // [N, K] f16
                      float* __restrict__ C, int ldc,
                      const float* __restrict__ bias,
                      const float* __restrict__ res,
                      const float* __restrict__ bn_g, const float* __restrict__ bn_b,
                      const float* __restrict__ bn_m, const float* __restrict__ bn_v,
                      int M, int N, int K)
{
    const int lane  = threadIdx.x & 31;
    const int wave  = threadIdx.x >> 5;
    const int mtile = M >> 4;
    const int ntile = N >> 4;
    const int tile  = blockIdx.x * 8 + wave;
    if (tile >= mtile * ntile) return;                  // wave-uniform exit

    const int tm   = tile / ntile;
    const int tn   = tile % ntile;
    const int half = lane >> 4;                         // lane group: +0 / +8 in K
    const int l15  = lane & 15;
    const int arow = tm * 16 + l15;                     // A row owned by this lane
    const int bcol = tn * 16 + l15;                     // W col owned by this lane

    // lane base pointers; fragment K pattern = {k0+h8+0..7, k0+16+h8+0..7}
    const float*    ap = A  + (long)arow * lda + half * 8;
    const _Float16* bp = Wt + (long)bcol * K   + half * 8;

    const int K32 = K & ~31;
    v8f acc = {};

    for (int k0 = 0; k0 < K32; k0 += 32) {
        float4 a0 = *(const float4*)(ap + k0);
        float4 a1 = *(const float4*)(ap + k0 + 4);
        float4 a2 = *(const float4*)(ap + k0 + 16);
        float4 a3 = *(const float4*)(ap + k0 + 20);
        v8h    b0 = *(const v8h*)(bp + k0);
        v8h    b1 = *(const v8h*)(bp + k0 + 16);

        v16h af, bf;
        af[0]  = (_Float16)a0.x; af[1]  = (_Float16)a0.y;
        af[2]  = (_Float16)a0.z; af[3]  = (_Float16)a0.w;
        af[4]  = (_Float16)a1.x; af[5]  = (_Float16)a1.y;
        af[6]  = (_Float16)a1.z; af[7]  = (_Float16)a1.w;
        af[8]  = (_Float16)a2.x; af[9]  = (_Float16)a2.y;
        af[10] = (_Float16)a2.z; af[11] = (_Float16)a2.w;
        af[12] = (_Float16)a3.x; af[13] = (_Float16)a3.y;
        af[14] = (_Float16)a3.z; af[15] = (_Float16)a3.w;
#pragma unroll
        for (int i = 0; i < 8; ++i) { bf[i] = b0[i]; bf[i + 8] = b1[i]; }

        acc = __builtin_amdgcn_wmma_f32_16x16x32_f16(
                  false, af, false, bf, (short)0, acc, false, false);
    }

    if (K32 < K) {                                      // only the K=3 GEMMs land here
        v16h af, bf;
#pragma unroll
        for (int i = 0; i < 16; ++i) {
            int kk = K32 + ((i < 8) ? (half * 8 + i) : (16 + half * 8 + (i - 8)));
            int kc = (kk < K) ? kk : (K - 1);           // clamp: address always valid
            float    av = A[(long)arow * lda + kc];
            _Float16 bv = Wt[(long)bcol * K + kc];
            af[i] = (kk < K) ? (_Float16)av : (_Float16)0.0f;   // v_cndmask, no branch
            bf[i] = (kk < K) ? bv           : (_Float16)0.0f;
        }
        acc = __builtin_amdgcn_wmma_f32_16x16x32_f16(
                  false, af, false, bf, (short)0, acc, false, false);
    }

    const int col = tn * 16 + l15;
#pragma unroll
    for (int r = 0; r < 8; ++r) {
        int row = tm * 16 + r + half * 8;               // C layout: VGPR r -> M=r / r+8
        float v = acc[r];
        if (EPI != 0) v += bias[col];
        if (EPI == 2) {                                 // BatchNorm + ReLU
            float rs = rsqrtf(bn_v[col] + GEPS);
            v = (v - bn_m[col]) * rs * bn_g[col] + bn_b[col];
            v = fmaxf(v, 0.0f);
        } else if (EPI == 3) {                          // exact GELU
            v = 0.5f * v * (1.0f + erff(v * 0.70710678118654752f));
        } else if (EPI == 4) {                          // residual add
            v += res[(long)row * ldc + col];
        }
        C[(long)row * ldc + col] = v;
    }
}

// ---------------------------------------------------------------------------
// Fused softmax attention. qkv layout (B,N,3*d), channel = which*d + head*dh + e
// one block per (b, head, query row n)
// ---------------------------------------------------------------------------
__global__ __launch_bounds__(256)
void attention_kernel(const float* __restrict__ qkv, const float* __restrict__ addm,
                      float* __restrict__ out, int N, int d, int dh, float scale)
{
    __shared__ float att[512];
    __shared__ float qs[64];
    __shared__ float red[256];

    int idx  = blockIdx.x;
    int n    = idx % N;  idx /= N;
    int head = idx % NHEADS;
    int b    = idx / NHEADS;
    int tid  = threadIdx.x;

    const float* qrow = qkv + ((long)(b * N + n) * 3 * d) + head * dh;
    if (tid < dh) qs[tid] = qrow[tid];
    __syncthreads();

    float lmax = -3.0e38f;
    for (int m = tid; m < N; m += 256) {
        const float* krow = qkv + ((long)(b * N + m) * 3 * d) + d + head * dh;
        float s = 0.0f;
        for (int c = 0; c < dh; ++c) s += qs[c] * krow[c];
        s = s * scale + addm[((long)b * N + n) * N + m];
        att[m] = s;
        lmax = fmaxf(lmax, s);
    }
    red[tid] = lmax; __syncthreads();
    for (int st = 128; st > 0; st >>= 1) {
        if (tid < st) red[tid] = fmaxf(red[tid], red[tid + st]);
        __syncthreads();
    }
    float gmax = red[0]; __syncthreads();

    float lsum = 0.0f;
    for (int m = tid; m < N; m += 256) {
        float e = __expf(att[m] - gmax);
        att[m] = e;
        lsum += e;
    }
    red[tid] = lsum; __syncthreads();
    for (int st = 128; st > 0; st >>= 1) {
        if (tid < st) red[tid] += red[tid + st];
        __syncthreads();
    }
    float inv = 1.0f / red[0];
    __syncthreads();

    if (tid < dh) {
        float acc = 0.0f;
        for (int m = 0; m < N; ++m)
            acc += att[m] * qkv[((long)(b * N + m) * 3 * d) + 2 * d + head * dh + tid];
        out[((long)(b * N + n)) * d + head * dh + tid] = acc * inv;
    }
}

// ---------------------------------------------------------------------------
__global__ __launch_bounds__(256)
void layernorm_kernel(const float* __restrict__ x, const float* __restrict__ g,
                      const float* __restrict__ b, float* __restrict__ y, int d)
{
    __shared__ float red[256];
    long row = blockIdx.x;
    const float* xr = x + row * d;
    float s = 0.0f, s2 = 0.0f;
    for (int c = threadIdx.x; c < d; c += 256) { float v = xr[c]; s += v; s2 += v * v; }
    red[threadIdx.x] = s; __syncthreads();
    for (int st = 128; st > 0; st >>= 1) { if (threadIdx.x < st) red[threadIdx.x] += red[threadIdx.x + st]; __syncthreads(); }
    float mean = red[0] / d; __syncthreads();
    red[threadIdx.x] = s2; __syncthreads();
    for (int st = 128; st > 0; st >>= 1) { if (threadIdx.x < st) red[threadIdx.x] += red[threadIdx.x + st]; __syncthreads(); }
    float var = red[0] / d - mean * mean;
    float rs  = rsqrtf(var + GEPS);
    for (int c = threadIdx.x; c < d; c += 256)
        y[row * d + c] = (xr[c] - mean) * rs * g[c] + b[c];
}

__global__ void add_kernel(const float* a, const float* b, float* y, long n)
{
    long t = (long)blockIdx.x * blockDim.x + threadIdx.x;
    if (t < n) y[t] = a[t] + b[t];
}

// max over K then concat [broadcast(max), f] : (G*K, C) -> (G*K, 2C)
__global__ void maxcat_kernel(const float* __restrict__ f, float* __restrict__ out,
                              int Gc, int Kn, int C)
{
    long t = (long)blockIdx.x * blockDim.x + threadIdx.x;
    if (t >= (long)Gc * C) return;
    int g = (int)(t / C), c = (int)(t % C);
    float mx = -3.0e38f;
    for (int k = 0; k < Kn; ++k) mx = fmaxf(mx, f[((long)(g * Kn + k)) * C + c]);
    for (int k = 0; k < Kn; ++k) {
        long r = (long)(g * Kn + k);
        out[r * (2 * C) + c]     = mx;
        out[r * (2 * C) + C + c] = f[r * C + c];
    }
}

// max over K: (G*K, C) -> (G, C)
__global__ void maxpool_kernel(const float* __restrict__ f, float* __restrict__ out,
                               int Gc, int Kn, int C)
{
    long t = (long)blockIdx.x * blockDim.x + threadIdx.x;
    if (t >= (long)Gc * C) return;
    int g = (int)(t / C), c = (int)(t % C);
    float mx = -3.0e38f;
    for (int k = 0; k < Kn; ++k) mx = fmaxf(mx, f[((long)(g * Kn + k)) * C + c]);
    out[(long)g * C + c] = mx;
}

// ---------------------------------------------------------------------------
// Masking / ordering logic
// ---------------------------------------------------------------------------
__device__ inline float hash_uniform(unsigned x)
{
    x ^= x >> 16; x *= 0x7feb352dU;
    x ^= x >> 15; x *= 0x846ca68bU;
    x ^= x >> 16;
    return ((float)(x >> 8)) * (1.0f / 16777216.0f) + 1.0e-7f;
}

__global__ void mask_last_kernel(const float* __restrict__ sals, int* __restrict__ mask,
                                 int Bn, int G, int num_keep)
{
    int b = blockIdx.x * blockDim.x + threadIdx.x;
    if (b >= Bn) return;
    float sc[64]; int id[64];
    float mx = -3.0e38f;
    for (int j = 0; j < G; ++j) mx = fmaxf(mx, sals[b * G + j]);
    float sum = 0.0f;
    for (int j = 0; j < G; ++j) { sc[j] = __expf(sals[b * G + j] - mx); sum += sc[j]; }
    for (int j = 0; j < G; ++j) {
        float a = sc[j] / sum;
        float u = hash_uniform((unsigned)(b * G + j) * 2654435761u + 12345u);
        u = fminf(fmaxf(u, 1e-6f), 1.0f - 1e-6f);
        float gum = -__logf(-__logf(u));
        sc[j] = __logf(a + 1e-12f) + gum;
        id[j] = j;
    }
    for (int i = 1; i < G; ++i) {                 // insertion sort, descending
        float key = sc[i]; int kid = id[i]; int j = i - 1;
        while (j >= 0 && sc[j] < key) { sc[j + 1] = sc[j]; id[j + 1] = id[j]; --j; }
        sc[j + 1] = key; id[j + 1] = kid;
    }
    for (int r = 0; r < G; ++r) mask[b * G + id[r]] = (r >= num_keep) ? 1 : 0;
}

__global__ void fill_int_kernel(int* p, int val, long n)
{
    long t = (long)blockIdx.x * blockDim.x + threadIdx.x;
    if (t < n) p[t] = val;
}

__global__ void propagate_mask_kernel(const int* __restrict__ mask_cur,
                                      const int* __restrict__ idx,
                                      int* __restrict__ mask_prev,
                                      int Bn, int g, int k)
{
    long t = (long)blockIdx.x * blockDim.x + threadIdx.x;
    if (t >= (long)Bn * g * k) return;
    long bg  = t / k;
    int  vis = (mask_cur[bg] == 0);
    int  j   = vis ? idx[t] : 0;                  // masked -> index 0 (ref quirk)
    mask_prev[j] = 0;
}

__global__ void order_kernel(const int* __restrict__ mask, int* __restrict__ order,
                             int* __restrict__ rank, float* __restrict__ vf,
                             int Bn, int G)
{
    int b = blockIdx.x * blockDim.x + threadIdx.x;
    if (b >= Bn) return;
    int pos = 0;
    for (int j = 0; j < G; ++j) if (mask[b * G + j] == 0) order[b * G + pos++] = j;
    int Lv = pos;
    for (int j = 0; j < G; ++j) if (mask[b * G + j] != 0) order[b * G + pos++] = j;
    for (int j = 0; j < G; ++j) rank[b * G + order[b * G + j]] = j;
    for (int j = 0; j < G; ++j) vf[b * G + j] = (j < Lv) ? 1.0f : 0.0f;
}

__global__ void gather_mul_kernel(const float* __restrict__ src, const int* __restrict__ order,
                                  const float* __restrict__ vf, float* __restrict__ dst,
                                  int Bn, int G, int C)
{
    long t = (long)blockIdx.x * blockDim.x + threadIdx.x;
    if (t >= (long)Bn * G * C) return;
    int c = (int)(t % C); long bg = t / C; int j = (int)(bg % G); int b = (int)(bg / G);
    int sj = order[b * G + j];
    dst[t] = src[((long)b * G + sj) * C + c] * vf[b * G + j];
}

__global__ void addmask_kernel(const float* __restrict__ mc, const float* __restrict__ vf,
                               float* __restrict__ addm, int Bn, int G, float R)
{
    long t = (long)blockIdx.x * blockDim.x + threadIdx.x;
    if (t >= (long)Bn * G * G) return;
    int m = (int)(t % G); long r = t / G; int n = (int)(r % G); int b = (int)(r / G);
    const float* pn = mc + ((long)b * G + n) * 3;
    const float* pm = mc + ((long)b * G + m) * 3;
    float dx = pn[0] - pm[0], dy = pn[1] - pm[1], dz = pn[2] - pm[2];
    float dist = sqrtf(fmaxf(dx * dx + dy * dy + dz * dz, 0.0f));
    float mv = 1.0f - vf[b * G + n] * vf[b * G + m];
    float am = (dist >= R) ? mv : 0.0f;
    addm[t] = (am > 0.0f) ? -1.0e9f : 0.0f;
}

__global__ void gather_rows_kernel(const float* __restrict__ src, const int* __restrict__ idx,
                                   float* __restrict__ dst, long nRows, int C)
{
    long t = (long)blockIdx.x * blockDim.x + threadIdx.x;
    if (t >= nRows * C) return;
    int c = (int)(t % C); long r = t / C;
    dst[t] = src[(long)idx[r] * C + c];
}

__global__ void compose_vis_kernel(const float* __restrict__ xv, const float* __restrict__ git,
                                   const int* __restrict__ mask, const int* __restrict__ rank,
                                   float* __restrict__ out, int Bn, int G, int C)
{
    long t = (long)blockIdx.x * blockDim.x + threadIdx.x;
    if (t >= (long)Bn * G * C) return;
    int c = (int)(t % C); long bg = t / C; int tt = (int)(bg % G); int b = (int)(bg / G);
    if (mask[b * G + tt] == 0)
        out[t] = xv[((long)b * G + rank[b * G + tt]) * C + c];
    else
        out[t] = git[t];
}

// ---------------------------------------------------------------------------
// Host-side orchestration
// ---------------------------------------------------------------------------
static inline int cdiv(int a, int b) { return (a + b - 1) / b; }

// transpose W (K,N) f32 -> Wt (N,K) f16, then WMMA GEMM
static void gemm(hipStream_t s, int epi,
                 const float* A, int lda, const float* W, _Float16* Wt,
                 float* C, int ldc, const float* bias, const float* res,
                 const float* bg, const float* bb, const float* bm, const float* bv,
                 int M, int N, int K)
{
    long wn = (long)K * N;
    transpose_h_kernel<<<dim3((unsigned)((wn + 255) / 256)), 256, 0, s>>>(W, Wt, K, N);
    int tiles = (M >> 4) * (N >> 4);
    dim3 grid(cdiv(tiles, 8)), blk(256);
    switch (epi) {
    case 0: wmma_gemm_kernel<0><<<grid, blk, 0, s>>>(A, lda, Wt, C, ldc, bias, res, bg, bb, bm, bv, M, N, K); break;
    case 1: wmma_gemm_kernel<1><<<grid, blk, 0, s>>>(A, lda, Wt, C, ldc, bias, res, bg, bb, bm, bv, M, N, K); break;
    case 2: wmma_gemm_kernel<2><<<grid, blk, 0, s>>>(A, lda, Wt, C, ldc, bias, res, bg, bb, bm, bv, M, N, K); break;
    case 3: wmma_gemm_kernel<3><<<grid, blk, 0, s>>>(A, lda, Wt, C, ldc, bias, res, bg, bb, bm, bv, M, N, K); break;
    case 4: wmma_gemm_kernel<4><<<grid, blk, 0, s>>>(A, lda, Wt, C, ldc, bias, res, bg, bb, bm, bv, M, N, K); break;
    }
}

extern "C" void kernel_launch(void* const* d_in, const int* in_sizes, int n_in,
                              void* d_out, int out_size, void* d_ws, size_t ws_size,
                              hipStream_t stream)
{
    (void)in_sizes; (void)n_in; (void)out_size; (void)ws_size;

    const int   Bz      = 16;
    const int   Gs[3]   = {512, 256, 64};
    const int   Ds[3]   = {96, 192, 384};
    const int   KNs[3]  = {16, 8, 8};
    const float Rs[3]   = {0.32f, 0.64f, 1.28f};
    const int   Ls      = 5;

    // ----- inputs (setup_inputs dict order; params pytree flattened) -----
    const float* neigh0  = (const float*)d_in[0];
    const float* centers[3] = { (const float*)d_in[3], (const float*)d_in[4], (const float*)d_in[5] };
    const int*   idx1    = (const int*)d_in[6];
    const int*   idx2    = (const int*)d_in[7];
    const float* sals    = (const float*)d_in[8];
    // params: 3 stages x {blk(11), norm(2), pos(4), te(16)} alphabetically
    auto P = [&](int stage, int leaf) -> const float* {
        return (const float*)d_in[9 + stage * 33 + leaf];
    };
    // blk: 0 Wm1, 1 Wm2, 2 Wproj, 3 Wqkv, 4 bm1, 5 bm2, 6 bproj,
    //      7 ln1_b, 8 ln1_g, 9 ln2_b, 10 ln2_g
    // norm: 11 b, 12 g;  pos: 13 W1, 14 W2, 15 b1, 16 b2
    // te: 17 W1, 18 W2, 19 W3, 20 W4, 21 b1, 22 b2, 23 b3, 24 b4,
    //     25 bn1_b, 26 bn1_g, 27 bn1_m, 28 bn1_v, 29 bn2_b, 30 bn2_g, 31 bn2_m, 32 bn2_v

    float* outp = (float*)d_out;
    const long out_off[3] = {0, (long)16 * 512 * 96, (long)16 * 512 * 96 + (long)16 * 256 * 192};

    // ----- workspace bump allocator -----
    char* wsp = (char*)d_ws;
    size_t off = 0;
    auto alloc = [&](size_t bytes) -> void* {
        off = (off + 255) & ~(size_t)255;
        void* p = wsp + off;
        off += bytes;
        return p;
    };
    int*      mask2 = (int*)alloc(sizeof(int) * Bz * 64);
    int*      mask1 = (int*)alloc(sizeof(int) * Bz * 256);
    int*      mask0 = (int*)alloc(sizeof(int) * Bz * 512);
    int*      order = (int*)alloc(sizeof(int) * Bz * 512);
    int*      rank  = (int*)alloc(sizeof(int) * Bz * 512);
    float*    vf    = (float*)alloc(sizeof(float) * Bz * 512);
    float*    git   = (float*)alloc(sizeof(float) * 786432);
    float*    xv    = (float*)alloc(sizeof(float) * 786432);
    float*    mcb   = (float*)alloc(sizeof(float) * Bz * 512 * 3);
    float*    addm  = (float*)alloc(sizeof(float) * (size_t)Bz * 512 * 512);
    float*    posb  = (float*)alloc(sizeof(float) * 786432);
    float*    hb    = (float*)alloc(sizeof(float) * 786432);
    float*    ab    = (float*)alloc(sizeof(float) * 786432);
    float*    qkvb  = (float*)alloc(sizeof(float) * 2359296);
    float*    atto  = (float*)alloc(sizeof(float) * 786432);
    float*    midb  = (float*)alloc(sizeof(float) * 3145728);
    float*    xvfA  = (float*)alloc(sizeof(float) * 786432);
    float*    xvfB  = (float*)alloc(sizeof(float) * 786432);
    float*    tbA   = (float*)alloc(sizeof(float) * 4194304);
    float*    tbB   = (float*)alloc(sizeof(float) * 4194304);
    _Float16* wT    = (_Float16*)alloc(sizeof(_Float16) * 589824);   // max K*N (Wm1/Wm2)

    auto egrid = [](long n) { return dim3((unsigned)((n + 255) / 256)); };

    // ===== masking chain =====
    mask_last_kernel<<<1, 32, 0, stream>>>(sals, mask2, Bz, 64, 13);           // num_keep=13
    fill_int_kernel<<<egrid(Bz * 256), 256, 0, stream>>>(mask1, 1, Bz * 256);
    propagate_mask_kernel<<<egrid((long)Bz * 64 * 8), 256, 0, stream>>>(mask2, idx2, mask1, Bz, 64, 8);
    fill_int_kernel<<<egrid(Bz * 512), 256, 0, stream>>>(mask0, 1, Bz * 512);
    propagate_mask_kernel<<<egrid((long)Bz * 256 * 8), 256, 0, stream>>>(mask1, idx1, mask0, Bz, 256, 8);
    int* masks[3] = {mask0, mask1, mask2};

    // ===== stages =====
    for (int i = 0; i < 3; ++i) {
        const int G  = Gs[i];
        const int d  = Ds[i];
        const int kn = KNs[i];
        const int dh = d / NHEADS;
        const float scale = 1.0f / sqrtf((float)dh);
        const long  BG = (long)Bz * G;

        const float* teW1 = P(i, 17); const float* teW2 = P(i, 18);
        const float* teW3 = P(i, 19); const float* teW4 = P(i, 20);
        const float* teb1 = P(i, 21); const float* teb2 = P(i, 22);
        const float* teb3 = P(i, 23); const float* teb4 = P(i, 24);
        const float* bn1b = P(i, 25); const float* bn1g = P(i, 26);
        const float* bn1m = P(i, 27); const float* bn1v = P(i, 28);
        const float* bn2b = P(i, 29); const float* bn2g = P(i, 30);
        const float* bn2m = P(i, 31); const float* bn2v = P(i, 32);
        const int Cin = (i == 0) ? 3 : Ds[i - 1];

        // ---- token embed (per-batch chunks) ----
        for (int b = 0; b < Bz; ++b) {
            int rows = G * kn;
            const float* Ain;
            if (i == 0) {
                Ain = neigh0 + (long)b * 512 * 16 * 3;
            } else {
                const float* src = (i == 1) ? xvfA : xvfB;
                const int* idxp = ((i == 1) ? idx1 : idx2) + (long)b * G * kn;
                gather_rows_kernel<<<egrid((long)rows * Cin), 256, 0, stream>>>(src, idxp, tbA, rows, Cin);
                Ain = tbA;
            }
            gemm(stream, 2, Ain, Cin, teW1, wT, tbB, 128, teb1, nullptr, bn1g, bn1b, bn1m, bn1v, rows, 128, Cin);
            gemm(stream, 1, tbB, 128, teW2, wT, tbA, 256, teb2, nullptr, nullptr, nullptr, nullptr, nullptr, rows, 256, 128);
            maxcat_kernel<<<egrid((long)G * 256), 256, 0, stream>>>(tbA, tbB, G, kn, 256);
            gemm(stream, 2, tbB, 512, teW3, wT, tbA, 512, teb3, nullptr, bn2g, bn2b, bn2m, bn2v, rows, 512, 512);
            gemm(stream, 1, tbA, 512, teW4, wT, tbB, d, teb4, nullptr, nullptr, nullptr, nullptr, nullptr, rows, d, 512);
            maxpool_kernel<<<egrid((long)G * d), 256, 0, stream>>>(tbB, git + (long)b * G * d, G, kn, d);
        }

        // ---- visible-first reorder ----
        order_kernel<<<1, 32, 0, stream>>>(masks[i], order, rank, vf, Bz, G);
        gather_mul_kernel<<<egrid(BG * d), 256, 0, stream>>>(git, order, vf, xv, Bz, G, d);
        gather_mul_kernel<<<egrid(BG * 3), 256, 0, stream>>>(centers[i], order, vf, mcb, Bz, G, 3);
        addmask_kernel<<<egrid(BG * G), 256, 0, stream>>>(mcb, vf, addm, Bz, G, Rs[i]);

        // ---- positional MLP: gelu(mc@W1+b1)@W2+b2 ----
        gemm(stream, 3, mcb, 3, P(i, 13), wT, ab, d, P(i, 15), nullptr, nullptr, nullptr, nullptr, nullptr, (int)BG, d, 3);
        gemm(stream, 1, ab, d, P(i, 14), wT, posb, d, P(i, 16), nullptr, nullptr, nullptr, nullptr, nullptr, (int)BG, d, d);

        // ---- transformer blocks ----
        for (int l = 0; l < Ls; ++l) {
            const float* Wqkv  = P(i, 3) + (long)l * d * 3 * d;
            const float* Wproj = P(i, 2) + (long)l * d * d;
            const float* bproj = P(i, 6) + (long)l * d;
            const float* ln1g  = P(i, 8) + (long)l * d;
            const float* ln1b  = P(i, 7) + (long)l * d;
            const float* ln2g  = P(i, 10) + (long)l * d;
            const float* ln2b  = P(i, 9) + (long)l * d;
            const float* Wm1   = P(i, 0) + (long)l * d * 4 * d;
            const float* bm1   = P(i, 4) + (long)l * 4 * d;
            const float* Wm2   = P(i, 1) + (long)l * 4 * d * d;
            const float* bm2   = P(i, 5) + (long)l * d;

            add_kernel<<<egrid(BG * d), 256, 0, stream>>>(xv, posb, hb, BG * d);      // h = x + pos
            layernorm_kernel<<<(unsigned)BG, 256, 0, stream>>>(hb, ln1g, ln1b, ab, d);
            gemm(stream, 0, ab, d, Wqkv, wT, qkvb, 3 * d, nullptr, nullptr, nullptr, nullptr, nullptr, nullptr, (int)BG, 3 * d, d);
            attention_kernel<<<(unsigned)(Bz * NHEADS * G), 256, 0, stream>>>(qkvb, addm, atto, G, d, dh, scale);
            gemm(stream, 4, atto, d, Wproj, wT, hb, d, bproj, hb, nullptr, nullptr, nullptr, nullptr, (int)BG, d, d); // h += o@Wp+bp
            layernorm_kernel<<<(unsigned)BG, 256, 0, stream>>>(hb, ln2g, ln2b, ab, d);
            gemm(stream, 3, ab, d, Wm1, wT, midb, 4 * d, bm1, nullptr, nullptr, nullptr, nullptr, nullptr, (int)BG, 4 * d, d);
            gemm(stream, 4, midb, 4 * d, Wm2, wT, xv, d, bm2, hb, nullptr, nullptr, nullptr, nullptr, (int)BG, d, 4 * d); // x = h+mlp
        }

        // ---- final norm -> output ----
        layernorm_kernel<<<(unsigned)BG, 256, 0, stream>>>(xv, P(i, 12), P(i, 11), outp + out_off[i], d);

        // ---- scatter visible tokens back for next stage ----
        if (i < 2) {
            float* xvf = (i == 0) ? xvfA : xvfB;
            compose_vis_kernel<<<egrid(BG * d), 256, 0, stream>>>(xv, git, masks[i], rank, xvf, Bz, G, d);
        }
    }
}
